// QuantMambaWrapper_79182017069512
// MI455X (gfx1250) — compile-verified, matrix operand order checked by reading, not run on hardware
//
#include <hip/hip_runtime.h>
#include <hip/hip_bf16.h>

// ---------------- problem constants ----------------
#define NBATCH  4
#define SEQ_L   2048
#define DMODEL  512
#define DINNER  1024
#define DSTATE  16
#define DTRANK  64
#define CONVK   4
#define MROWS   (NBATCH * SEQ_L)          // 8192
#define NPROJ   (DTRANK + 2 * DSTATE)     // 96

// ---------------- WMMA types / helpers ----------------
typedef __attribute__((ext_vector_type(16))) __bf16 v16bf;
typedef __attribute__((ext_vector_type(8)))  float  v8f;

__device__ __forceinline__ unsigned short f2bf(float f) {
  unsigned int u = __float_as_uint(f);
  u += 0x7FFFu + ((u >> 16) & 1u);      // round-to-nearest-even
  return (unsigned short)(u >> 16);
}

// One 16-bit A/B fragment: 8 contiguous bf16 (K=kb..kb+7) -> VGPRs 0-3,
// 8 contiguous bf16 at +16 (K=kb+16..kb+23) -> VGPRs 4-7.
// Matches the CDNA5 16-bit 16x32 fragment layout (CDNA5 ISA §7.12.2).
__device__ __forceinline__ v16bf load_frag(const unsigned short* p) {
  union { v16bf v; uint4 q[2]; } u;
  u.q[0] = *(const uint4*)(p);
  u.q[1] = *(const uint4*)(p + 16);
  return u.v;
}

__device__ __forceinline__ v8f wmma_bf16(v16bf a, v16bf b, v8f c) {
  return __builtin_amdgcn_wmma_f32_16x16x32_bf16(false, a, false, b, (short)0, c,
                                                 false, false);
}

// C/D layout: VGPR v holds row m0+v (lanes 0-15) / m0+8+v (lanes 16-31),
// column n0 + (lane&15).
__device__ __forceinline__ void store_tile_f32(float* C, int ldc, int m0, int n0,
                                               v8f c, int lane) {
  int col  = n0 + (lane & 15);
  int row0 = m0 + ((lane >> 4) << 3);
#pragma unroll
  for (int v = 0; v < 8; ++v) C[(size_t)(row0 + v) * ldc + col] = c[v];
}

__device__ __forceinline__ void store_tile_bf16(unsigned short* C, int ldc, int m0,
                                                int n0, v8f c, int lane) {
  int col  = n0 + (lane & 15);
  int row0 = m0 + ((lane >> 4) << 3);
#pragma unroll
  for (int v = 0; v < 8; ++v) C[(size_t)(row0 + v) * ldc + col] = f2bf(c[v]);
}

__device__ __forceinline__ float softplusf(float x) {
  return (x > 20.f) ? x : log1pf(__expf(x));
}

// 2M x 4N accumulator block: 8 WMMA per K-step off 6 fragment loads.
struct Acc24 { v8f c[2][4]; };

__device__ __forceinline__ void kstep_2x4(const unsigned short* __restrict__ A,
                                          const unsigned short* __restrict__ B,
                                          size_t ra0, size_t ra1,
                                          const size_t* rb, int k, Acc24& acc) {
  v16bf a0 = load_frag(A + ra0 + k);
  v16bf a1 = load_frag(A + ra1 + k);
#pragma unroll
  for (int j = 0; j < 4; ++j) {
    v16bf b = load_frag(B + rb[j] + k);
    acc.c[0][j] = wmma_bf16(a0, b, acc.c[0][j]);
    acc.c[1][j] = wmma_bf16(a1, b, acc.c[1][j]);
  }
}

// ---------------- conversion kernels ----------------
__global__ void k_cvt_bf16(const float* __restrict__ in,
                           unsigned short* __restrict__ out, int n) {
  int i = blockIdx.x * blockDim.x + threadIdx.x;
  if (i < n) out[i] = f2bf(in[i]);
}

__global__ void k_neg_exp(const float* __restrict__ in, float* __restrict__ out,
                          int n) {
  int i = blockIdx.x * blockDim.x + threadIdx.x;
  if (i < n) out[i] = -__expf(in[i]);
}

// ---------------- GEMM 1: xz = x(flip?) @ W_in^T (M=8192,N=2048,K=512), 2x4 tile ----------------
// Output columns [0,1024) -> xb, [1024,2048) -> zb (split is 64-aligned).
// Backward direction reads x rows L-flipped so downstream buffers live in the
// scan time domain.
__global__ void gemm_inproj(const unsigned short* __restrict__ X,   // [MROWS][512] bf16
                            const unsigned short* __restrict__ W,   // [2048][512]  bf16
                            float* __restrict__ xb,                 // [MROWS][1024]
                            float* __restrict__ zb,                 // [MROWS][1024]
                            int flip) {
  const int K = DMODEL, NT = (2 * DINNER) / 64;
  int wave = (int)((blockIdx.x * blockDim.x + threadIdx.x) >> 5);
  int lane = (int)(threadIdx.x & 31);
  int m0 = (wave / NT) * 32, n0 = (wave % NT) * 64;
  int lr = lane & 15, kb = (lane >> 4) << 3;

  int mA0 = m0 + lr, mA1 = m0 + 16 + lr;
  if (flip) {
    mA0 = (mA0 & ~(SEQ_L - 1)) + (SEQ_L - 1 - (mA0 & (SEQ_L - 1)));
    mA1 = (mA1 & ~(SEQ_L - 1)) + (SEQ_L - 1 - (mA1 & (SEQ_L - 1)));
  }
  size_t ra0 = (size_t)mA0 * K + kb;
  size_t ra1 = (size_t)mA1 * K + kb;
  size_t rb[4];
#pragma unroll
  for (int j = 0; j < 4; ++j) rb[j] = (size_t)(n0 + 16 * j + lr) * K + kb;

  Acc24 acc = {};
  for (int k = 0; k < K; k += 32) {
    __builtin_prefetch(X + ra0 + k + 64, 0, 3);   // WGP-scope near prefetch
    __builtin_prefetch(W + rb[0] + k + 64, 0, 3);
    kstep_2x4(X, W, ra0, ra1, rb, k, acc);
  }
  float* dst = (n0 < DINNER) ? xb : zb;
  int nc = (n0 < DINNER) ? n0 : (n0 - DINNER);
#pragma unroll
  for (int j = 0; j < 4; ++j) {
    store_tile_f32(dst, DINNER, m0,      nc + 16 * j, acc.c[0][j], lane);
    store_tile_f32(dst, DINNER, m0 + 16, nc + 16 * j, acc.c[1][j], lane);
  }
}

// ---------------- depthwise conv (K=4, causal) + SiLU ----------------
__global__ void conv_silu(const float* __restrict__ xb,     // [MROWS][1024]
                          const float* __restrict__ cw,     // [1024][4]
                          const float* __restrict__ cb,     // [1024]
                          float* __restrict__ xcF,          // [MROWS][1024]
                          unsigned short* __restrict__ xcB) // bf16 copy for x-proj GEMM
{
  int i = blockIdx.x * blockDim.x + threadIdx.x;
  if (i >= MROWS * DINNER) return;
  int d = i & (DINNER - 1);
  int m = i >> 10;
  int t = m & (SEQ_L - 1);
  float acc = cb[d];
#pragma unroll
  for (int kk = 0; kk < CONVK; ++kk) {
    int tt = t - (CONVK - 1) + kk;
    if (tt >= 0)
      acc = fmaf(cw[d * CONVK + kk], xb[(size_t)(m - (CONVK - 1) + kk) * DINNER + d], acc);
  }
  float s = acc / (1.f + __expf(-acc));   // SiLU
  xcF[i] = s;
  xcB[i] = f2bf(s);
}

// ---------------- GEMM 2: proj = xconv @ Wx^T (M=8192,N=96,K=1024), 2x2 tile ----------------
__global__ void gemm_xproj(const unsigned short* __restrict__ Xc,  // [MROWS][1024] bf16
                           const unsigned short* __restrict__ W,   // [96][1024]    bf16
                           float* __restrict__ proj,               // [MROWS][96]
                           unsigned short* __restrict__ dtin)      // [MROWS][64] bf16
{
  const int K = DINNER, NT = NPROJ / 32;
  int wave = (int)((blockIdx.x * blockDim.x + threadIdx.x) >> 5);
  int lane = (int)(threadIdx.x & 31);
  int m0 = (wave / NT) * 32, n0 = (wave % NT) * 32;
  int lr = lane & 15, kb = (lane >> 4) << 3;

  const unsigned short* pa0 = Xc + (size_t)(m0 + lr) * K + kb;
  const unsigned short* pa1 = Xc + (size_t)(m0 + 16 + lr) * K + kb;
  const unsigned short* pb0 = W + (size_t)(n0 + lr) * K + kb;
  const unsigned short* pb1 = W + (size_t)(n0 + 16 + lr) * K + kb;

  v8f c00 = {}, c01 = {}, c10 = {}, c11 = {};
  for (int k = 0; k < K; k += 32) {
    __builtin_prefetch(pa0 + k + 64, 0, 3);
    v16bf a0 = load_frag(pa0 + k);
    v16bf a1 = load_frag(pa1 + k);
    v16bf b0 = load_frag(pb0 + k);
    v16bf b1 = load_frag(pb1 + k);
    c00 = wmma_bf16(a0, b0, c00);
    c01 = wmma_bf16(a0, b1, c01);
    c10 = wmma_bf16(a1, b0, c10);
    c11 = wmma_bf16(a1, b1, c11);
  }
  store_tile_f32(proj, NPROJ, m0,      n0,      c00, lane);
  store_tile_f32(proj, NPROJ, m0,      n0 + 16, c01, lane);
  store_tile_f32(proj, NPROJ, m0 + 16, n0,      c10, lane);
  store_tile_f32(proj, NPROJ, m0 + 16, n0 + 16, c11, lane);
  if (n0 < DTRANK) {   // dt_in columns -> also bf16 for the dt GEMM
    store_tile_bf16(dtin, DTRANK, m0,      n0,      c00, lane);
    store_tile_bf16(dtin, DTRANK, m0,      n0 + 16, c01, lane);
    store_tile_bf16(dtin, DTRANK, m0 + 16, n0,      c10, lane);
    store_tile_bf16(dtin, DTRANK, m0 + 16, n0 + 16, c11, lane);
  }
}

// ---------------- GEMM 3: dt = softplus(dtin @ Wdt^T + bdt) (M=8192,N=1024,K=64), 2x4 tile ----------------
__global__ void gemm_dt(const unsigned short* __restrict__ DtIn, // [MROWS][64] bf16
                        const unsigned short* __restrict__ W,    // [1024][64]  bf16
                        const float* __restrict__ bdt,           // [1024]
                        float* __restrict__ dt)                  // [MROWS][1024]
{
  const int K = DTRANK, NT = DINNER / 64;
  int wave = (int)((blockIdx.x * blockDim.x + threadIdx.x) >> 5);
  int lane = (int)(threadIdx.x & 31);
  int m0 = (wave / NT) * 32, n0 = (wave % NT) * 64;
  int lr = lane & 15, kb = (lane >> 4) << 3;

  size_t ra0 = (size_t)(m0 + lr) * K + kb;
  size_t ra1 = (size_t)(m0 + 16 + lr) * K + kb;
  size_t rb[4];
#pragma unroll
  for (int j = 0; j < 4; ++j) rb[j] = (size_t)(n0 + 16 * j + lr) * K + kb;

  Acc24 acc = {};
#pragma unroll
  for (int k = 0; k < K; k += 32) kstep_2x4(DtIn, W, ra0, ra1, rb, k, acc);

  int colL = lane & 15;
  int row0 = m0 + ((lane >> 4) << 3);
#pragma unroll
  for (int j = 0; j < 4; ++j) {
    int col = n0 + 16 * j + colL;
    float bv = bdt[col];
#pragma unroll
    for (int v = 0; v < 8; ++v) {
      dt[(size_t)(row0 + v) * DINNER + col]      = softplusf(acc.c[0][j][v] + bv);
      dt[(size_t)(row0 + 16 + v) * DINNER + col] = softplusf(acc.c[1][j][v] + bv);
    }
  }
}

// ---------------- selective scan + D skip + SiLU gate ----------------
// One wave = 2 channels (b,d); lane = state index s = lane&15.
// h[s] recurrence over t; y = sum_s h[s]*C[s] via 16-lane shfl_xor reduction.
// Backward direction stores gated output into the L-flipped row so the final
// out-GEMM needs no flip.
__global__ void scan_kernel(const float* __restrict__ dtF,   // [MROWS][1024]
                            const float* __restrict__ xcF,   // [MROWS][1024]
                            const float* __restrict__ proj,  // [MROWS][96] (B:64..79, C:80..95)
                            const float* __restrict__ zbF,   // [MROWS][1024]
                            const float* __restrict__ Aneg,  // [16][1024]  = -exp(A_log)
                            const float* __restrict__ Dp,    // [1024]
                            unsigned short* __restrict__ yg, // [MROWS][1024] bf16
                            int flip) {
  int wave = (int)((blockIdx.x * blockDim.x + threadIdx.x) >> 5);
  int lane = (int)(threadIdx.x & 31);
  int s = lane & 15;
  int ch = wave * 2 + (lane >> 4);      // 0..4095 = b*1024 + d
  int b = ch >> 10;
  int d = ch & (DINNER - 1);

  float Av = Aneg[s * DINNER + d];
  float Dd = Dp[d];
  float h = 0.f;
  size_t base = (size_t)b * SEQ_L;

  for (int t = 0; t < SEQ_L; ++t) {
    size_t m = base + t;
    float dtv = dtF[m * DINNER + d];
    float xv  = xcF[m * DINNER + d];
    float Bv  = proj[m * NPROJ + DTRANK + s];
    float Cv  = proj[m * NPROJ + DTRANK + DSTATE + s];
    float dA  = __expf(dtv * Av);
    h = fmaf(dA, h, dtv * xv * Bv);
    float y = h * Cv;
    y += __shfl_xor(y, 1);
    y += __shfl_xor(y, 2);
    y += __shfl_xor(y, 4);
    y += __shfl_xor(y, 8);
    if (s == 0) {
      float zv = zbF[m * DINNER + d];
      float gate = zv / (1.f + __expf(-zv));
      float o = (y + Dd * xv) * gate;
      size_t mo = base + (flip ? (SEQ_L - 1 - t) : t);
      yg[mo * DINNER + d] = f2bf(o);
    }
  }
}

// ---------------- GEMM 4: out = yg_f @ Wout_f^T + yg_b @ Wout_b^T, 2x4 tile ----------------
__global__ void gemm_out_dual(const unsigned short* __restrict__ Af, // [MROWS][1024] bf16
                              const unsigned short* __restrict__ Bf, // [512][1024]   bf16
                              const unsigned short* __restrict__ Ab,
                              const unsigned short* __restrict__ Bb,
                              float* __restrict__ Out)               // [MROWS][512]
{
  const int K = DINNER, NT = DMODEL / 64;
  int wave = (int)((blockIdx.x * blockDim.x + threadIdx.x) >> 5);
  int lane = (int)(threadIdx.x & 31);
  int m0 = (wave / NT) * 32, n0 = (wave % NT) * 64;
  int lr = lane & 15, kb = (lane >> 4) << 3;

  size_t ra0 = (size_t)(m0 + lr) * K + kb;
  size_t ra1 = (size_t)(m0 + 16 + lr) * K + kb;
  size_t rb[4];
#pragma unroll
  for (int j = 0; j < 4; ++j) rb[j] = (size_t)(n0 + 16 * j + lr) * K + kb;

  Acc24 acc = {};
#pragma unroll
  for (int pass = 0; pass < 2; ++pass) {
    const unsigned short* A = pass ? Ab : Af;
    const unsigned short* B = pass ? Bb : Bf;
    for (int k = 0; k < K; k += 32) {
      __builtin_prefetch(A + ra0 + k + 64, 0, 3);
      __builtin_prefetch(B + rb[0] + k + 64, 0, 3);
      kstep_2x4(A, B, ra0, ra1, rb, k, acc);
    }
  }
#pragma unroll
  for (int j = 0; j < 4; ++j) {
    store_tile_f32(Out, DMODEL, m0,      n0 + 16 * j, acc.c[0][j], lane);
    store_tile_f32(Out, DMODEL, m0 + 16, n0 + 16 * j, acc.c[1][j], lane);
  }
}

// ---------------- host launch ----------------
extern "C" void kernel_launch(void* const* d_in, const int* in_sizes, int n_in,
                              void* d_out, int out_size, void* d_ws, size_t ws_size,
                              hipStream_t stream) {
  const float* x = (const float*)d_in[0];
  const float* W_in[2]  = {(const float*)d_in[1],  (const float*)d_in[10]};
  const float* conv_w[2]= {(const float*)d_in[2],  (const float*)d_in[11]};
  const float* conv_b[2]= {(const float*)d_in[3],  (const float*)d_in[12]};
  const float* A_log[2] = {(const float*)d_in[4],  (const float*)d_in[13]};
  const float* Dp[2]    = {(const float*)d_in[5],  (const float*)d_in[14]};
  const float* Wx[2]    = {(const float*)d_in[6],  (const float*)d_in[15]};
  const float* Wdt[2]   = {(const float*)d_in[7],  (const float*)d_in[16]};
  const float* bdt[2]   = {(const float*)d_in[8],  (const float*)d_in[17]};
  const float* Wout[2]  = {(const float*)d_in[9],  (const float*)d_in[18]};

  char* cur = (char*)d_ws;
  auto alloc = [&](size_t bytes) -> void* {
    void* p = (void*)cur;
    cur += (bytes + 255) & ~(size_t)255;
    return p;
  };

  // persistent bf16 copies
  unsigned short* xBf = (unsigned short*)alloc((size_t)MROWS * DMODEL * 2);
  unsigned short *WinB[2], *WxB[2], *WdtB[2], *WoutB[2], *ygB[2];
  float* AnegB[2];
  for (int i = 0; i < 2; ++i) {
    WinB[i]  = (unsigned short*)alloc((size_t)2 * DINNER * DMODEL * 2);
    WxB[i]   = (unsigned short*)alloc((size_t)NPROJ * DINNER * 2);
    WdtB[i]  = (unsigned short*)alloc((size_t)DINNER * DTRANK * 2);
    WoutB[i] = (unsigned short*)alloc((size_t)DMODEL * DINNER * 2);
    AnegB[i] = (float*)alloc((size_t)DSTATE * DINNER * 4);
    ygB[i]   = (unsigned short*)alloc((size_t)MROWS * DINNER * 2);
  }
  // per-direction scratch, reused between directions (sequential on stream)
  float* xbF            = (float*)alloc((size_t)MROWS * DINNER * 4);
  float* zbF            = (float*)alloc((size_t)MROWS * DINNER * 4);
  float* xcF            = (float*)alloc((size_t)MROWS * DINNER * 4);
  unsigned short* xcB   = (unsigned short*)alloc((size_t)MROWS * DINNER * 2);
  float* projF          = (float*)alloc((size_t)MROWS * NPROJ * 4);
  unsigned short* dtinB = (unsigned short*)alloc((size_t)MROWS * DTRANK * 2);
  float* dtF            = (float*)alloc((size_t)MROWS * DINNER * 4);

  const int TB = 256;
  auto nb = [](int n, int tb) { return (n + tb - 1) / tb; };

  // one-time conversions
  k_cvt_bf16<<<nb(MROWS * DMODEL, TB), TB, 0, stream>>>(x, xBf, MROWS * DMODEL);
  for (int i = 0; i < 2; ++i) {
    k_cvt_bf16<<<nb(2 * DINNER * DMODEL, TB), TB, 0, stream>>>(W_in[i], WinB[i], 2 * DINNER * DMODEL);
    k_cvt_bf16<<<nb(NPROJ * DINNER, TB), TB, 0, stream>>>(Wx[i], WxB[i], NPROJ * DINNER);
    k_cvt_bf16<<<nb(DINNER * DTRANK, TB), TB, 0, stream>>>(Wdt[i], WdtB[i], DINNER * DTRANK);
    k_cvt_bf16<<<nb(DMODEL * DINNER, TB), TB, 0, stream>>>(Wout[i], WoutB[i], DMODEL * DINNER);
    k_neg_exp<<<nb(DSTATE * DINNER, TB), TB, 0, stream>>>(A_log[i], AnegB[i], DSTATE * DINNER);
  }

  // per-direction pipeline (dir 1 reuses the shared f32 scratch)
  for (int dir = 0; dir < 2; ++dir) {
    int g1 = (MROWS / 32) * ((2 * DINNER) / 64) * 32 / TB;   // 1024 blocks
    gemm_inproj<<<g1, TB, 0, stream>>>(xBf, WinB[dir], xbF, zbF, dir);

    conv_silu<<<nb(MROWS * DINNER, TB), TB, 0, stream>>>(xbF, conv_w[dir], conv_b[dir], xcF, xcB);

    int g2 = (MROWS / 32) * (NPROJ / 32) * 32 / TB;          // 96 blocks
    gemm_xproj<<<g2, TB, 0, stream>>>(xcB, WxB[dir], projF, dtinB);

    int g3 = (MROWS / 32) * (DINNER / 64) * 32 / TB;         // 512 blocks
    gemm_dt<<<g3, TB, 0, stream>>>(dtinB, WdtB[dir], bdt[dir], dtF);

    int g5 = (NBATCH * DINNER / 2) * 32 / TB;                // 256 blocks
    scan_kernel<<<g5, TB, 0, stream>>>(dtF, xcF, projF, zbF, AnegB[dir], Dp[dir], ygB[dir], dir);
  }

  int g4 = (MROWS / 32) * (DMODEL / 64) * 32 / TB;           // 256 blocks
  gemm_out_dual<<<g4, TB, 0, stream>>>(ygB[0], WoutB[0], ygB[1], WoutB[1], (float*)d_out);

  (void)in_sizes; (void)n_in; (void)out_size; (void)ws_size;
}
